// FeatureFusionModule_5772436045949
// MI455X (gfx1250) — compile-verified, matrix operand order checked by reading, not run on hardware
//
#include <hip/hip_runtime.h>

#define HDIM 1024
#define BATCH 8192

typedef __attribute__((ext_vector_type(8)))  __bf16 v8bf;
typedef __attribute__((ext_vector_type(16))) __bf16 v16bf;
typedef __attribute__((ext_vector_type(8)))  float  v8f;

typedef int v4i __attribute__((vector_size(16)));
typedef __attribute__((address_space(1))) v4i gv4i;   // global 128-bit chunk
typedef __attribute__((address_space(3))) v4i lv4i;   // LDS 128-bit chunk

#if defined(__gfx1250__) && __has_builtin(__builtin_amdgcn_global_load_async_to_lds_b128)
#define USE_ASYNC 1
#else
#define USE_ASYNC 0
#endif

#if USE_ASYNC
#if __has_builtin(__builtin_amdgcn_s_wait_asynccnt)
#define WAIT_ASYNC() __builtin_amdgcn_s_wait_asynccnt(0)
#else
#define WAIT_ASYNC() asm volatile("s_wait_asynccnt 0" ::: "memory")
#endif
#endif

// ---------------------------------------------------------------------------
// fp32 -> bf16 conversion (weights / activations), 8 elems per thread.
// ---------------------------------------------------------------------------
__global__ __launch_bounds__(256)
void k_cvt(const float* __restrict__ s, __bf16* __restrict__ d, size_t n)
{
    size_t i = ((size_t)blockIdx.x * 256 + threadIdx.x) * 8;
    if (i + 8 > n) return;
    float4 a = *(const float4*)(s + i);
    float4 b = *(const float4*)(s + i + 4);
    __bf16* dd = d + i;
    dd[0] = (__bf16)a.x; dd[1] = (__bf16)a.y; dd[2] = (__bf16)a.z; dd[3] = (__bf16)a.w;
    dd[4] = (__bf16)b.x; dd[5] = (__bf16)b.y; dd[6] = (__bf16)b.z; dd[7] = (__bf16)b.w;
}

// ---------------------------------------------------------------------------
// GEMM: C[M,N] = A[M,K] * W[N,K]^T + bias  (optional ReLU)
// A, W bf16 in memory. Block tile 128x128, K-step 32, double-buffered LDS
// staged with async global->LDS (ASYNCcnt) when available. 8 waves (4Mx2N);
// each wave owns 32x64 = 2x4 WMMA fragments -> 8 v_wmma per K-step.
// Outputs: fp32 Cf and/or bf16 Ch (either may be null).
// ---------------------------------------------------------------------------
__global__ __launch_bounds__(256, 2)
void k_gemm(const __bf16* __restrict__ A, const __bf16* __restrict__ W,
            const float* __restrict__ bias, float* __restrict__ Cf,
            __bf16* __restrict__ Ch, int M, int N, int K, int relu)
{
    __shared__ __align__(16) __bf16 As[2][128][40];   // 128x32 tiles, +8 pad
    __shared__ __align__(16) __bf16 Bs[2][128][40];

    const int tid  = threadIdx.x;
    const int lane = tid & 31;
    const int wave = tid >> 5;
    const int wm   = wave >> 1;        // 0..3 (M direction)
    const int wn   = wave & 1;         // 0..1 (N direction)
    const int half = lane >> 4;        // 0/1
    const int lr   = lane & 15;

    const int m0 = blockIdx.y * 128;
    const int n0 = blockIdx.x * 128;

    // staging: each tile is 128 rows x 32 bf16; thread -> 16 contiguous bf16
    const int sr = tid >> 1;           // 0..127
    const int sc = (tid & 1) << 4;     // 0 or 16
    const __bf16* ag = A + (size_t)(m0 + sr) * K + sc;
    const __bf16* bg = W + (size_t)(n0 + sr) * K + sc;

    auto stage = [&](int buf, int kt) {
#if USE_ASYNC
        __builtin_amdgcn_global_load_async_to_lds_b128(
            (gv4i*)(void*)(ag + kt),     (lv4i*)(void*)&As[buf][sr][sc],     0, 0);
        __builtin_amdgcn_global_load_async_to_lds_b128(
            (gv4i*)(void*)(ag + kt + 8), (lv4i*)(void*)&As[buf][sr][sc + 8], 0, 0);
        __builtin_amdgcn_global_load_async_to_lds_b128(
            (gv4i*)(void*)(bg + kt),     (lv4i*)(void*)&Bs[buf][sr][sc],     0, 0);
        __builtin_amdgcn_global_load_async_to_lds_b128(
            (gv4i*)(void*)(bg + kt + 8), (lv4i*)(void*)&Bs[buf][sr][sc + 8], 0, 0);
#else
        v8bf a0 = *(const v8bf*)(ag + kt);
        v8bf a1 = *(const v8bf*)(ag + kt + 8);
        v8bf b0 = *(const v8bf*)(bg + kt);
        v8bf b1 = *(const v8bf*)(bg + kt + 8);
        *(v8bf*)&As[buf][sr][sc]     = a0;
        *(v8bf*)&As[buf][sr][sc + 8] = a1;
        *(v8bf*)&Bs[buf][sr][sc]     = b0;
        *(v8bf*)&Bs[buf][sr][sc + 8] = b1;
#endif
    };

    // fragment loader (ISA 16-bit layout: lanes 0-15 K=0..7,16..23;
    // lanes 16-31 K=8..15,24..31)
    auto loadfrag = [&](const __bf16* rowp) -> v16bf {
        v8bf lo = *(const v8bf*)(rowp + half * 8);
        v8bf hi = *(const v8bf*)(rowp + 16 + half * 8);
        return __builtin_shufflevector(lo, hi,
            0,1,2,3,4,5,6,7,8,9,10,11,12,13,14,15);
    };

    v8f acc[2][4] = {};
    const int nk = K >> 5;

    stage(0, 0);
    for (int t = 0; t < nk; ++t) {
        const int cur = t & 1;
#if USE_ASYNC
        WAIT_ASYNC();
#endif
        __syncthreads();
        if (t + 1 < nk) stage(cur ^ 1, (t + 1) << 5);

        v16bf af[2], bfr[4];
#pragma unroll
        for (int mi = 0; mi < 2; mi++)
            af[mi] = loadfrag(&As[cur][wm * 32 + mi * 16 + lr][0]);
#pragma unroll
        for (int ni = 0; ni < 4; ni++)
            bfr[ni] = loadfrag(&Bs[cur][wn * 64 + ni * 16 + lr][0]);

#pragma unroll
        for (int mi = 0; mi < 2; mi++)
#pragma unroll
            for (int ni = 0; ni < 4; ni++)
                acc[mi][ni] = __builtin_amdgcn_wmma_f32_16x16x32_bf16(
                    false, af[mi], false, bfr[ni],
                    (short)0, acc[mi][ni], false, false);
        __syncthreads();
    }

    // epilogue: C layout VGPR j -> M=j (+8 for lane half 1), N=lr
#pragma unroll
    for (int ni = 0; ni < 4; ni++) {
        int col = n0 + wn * 64 + ni * 16 + lr;
        float bv = bias[col];
#pragma unroll
        for (int mi = 0; mi < 2; mi++) {
            int rbase = m0 + wm * 32 + mi * 16 + half * 8;
#pragma unroll
            for (int j = 0; j < 8; j++) {
                float tval = acc[mi][ni][j] + bv;
                if (relu) tval = fmaxf(tval, 0.f);
                size_t idx = (size_t)(rbase + j) * N + col;
                if (Cf) Cf[idx] = tval;
                if (Ch) Ch[idx] = (__bf16)tval;
            }
        }
    }
}

// ---------------------------------------------------------------------------
// Fused residual-add + LayerNorm; optional ReLU, optional bf16 copy,
// optional accumulate. One block (256 threads) per row of H=1024.
// ---------------------------------------------------------------------------
__global__ __launch_bounds__(256)
void k_addln(const float* __restrict__ x, const float* __restrict__ y,
             const float* __restrict__ g, const float* __restrict__ b,
             float* __restrict__ of, __bf16* __restrict__ oh,
             float* __restrict__ acc, int relu)
{
    __shared__ float red[16];
    const size_t row = blockIdx.x;
    const float* xr = x + row * HDIM;
    const float* yr = y ? y + row * HDIM : nullptr;

    float vals[4];
    float s = 0.f, s2 = 0.f;
#pragma unroll
    for (int i = 0; i < 4; i++) {
        int h = threadIdx.x + i * 256;
        float t = xr[h] + (yr ? yr[h] : 0.f);
        vals[i] = t; s += t; s2 += t * t;
    }
#pragma unroll
    for (int off = 16; off > 0; off >>= 1) {
        s  += __shfl_down(s, off, 32);
        s2 += __shfl_down(s2, off, 32);
    }
    int lane = threadIdx.x & 31, wv = threadIdx.x >> 5;
    if (lane == 0) { red[wv] = s; red[8 + wv] = s2; }
    __syncthreads();
    if (threadIdx.x == 0) {
        float a = 0.f, c = 0.f;
        for (int i = 0; i < 8; i++) { a += red[i]; c += red[8 + i]; }
        float mean = a * (1.f / HDIM);
        float var  = c * (1.f / HDIM) - mean * mean;
        red[0] = mean;
        red[1] = rsqrtf(var + 1e-5f);
    }
    __syncthreads();
    float mean = red[0], rstd = red[1];
#pragma unroll
    for (int i = 0; i < 4; i++) {
        int h = threadIdx.x + i * 256;
        float t = (vals[i] - mean) * rstd * g[h] + b[h];
        if (relu) t = fmaxf(t, 0.f);
        if (of)  of[row * HDIM + h] = t;
        if (oh)  oh[row * HDIM + h] = (__bf16)t;
        if (acc) acc[row * HDIM + h] += t;
    }
}

// ---------------------------------------------------------------------------
// 3-way attention fusion: logits = concat(e0,e1,e2) @ Watt^T + batt,
// softmax over 3, fused = sum_m w_m * e_m (written bf16 for the final GEMM).
// ---------------------------------------------------------------------------
__global__ __launch_bounds__(256)
void k_fuse(const float* __restrict__ e0, const float* __restrict__ e1,
            const float* __restrict__ e2, const float* __restrict__ Watt,
            const float* __restrict__ batt, __bf16* __restrict__ o)
{
    __shared__ float red[24];
    __shared__ float w[3];
    const size_t row = blockIdx.x;
    const float* r0 = e0 + row * HDIM;
    const float* r1 = e1 + row * HDIM;
    const float* r2 = e2 + row * HDIM;

    float d0 = 0.f, d1 = 0.f, d2 = 0.f;
#pragma unroll
    for (int i = 0; i < 4; i++) {
        int h = threadIdx.x + i * 256;
        float a = r0[h], bb = r1[h], c = r2[h];
        d0 += a * Watt[h]          + bb * Watt[HDIM + h]   + c * Watt[2*HDIM + h];
        d1 += a * Watt[3*HDIM + h] + bb * Watt[4*HDIM + h] + c * Watt[5*HDIM + h];
        d2 += a * Watt[6*HDIM + h] + bb * Watt[7*HDIM + h] + c * Watt[8*HDIM + h];
    }
#pragma unroll
    for (int off = 16; off > 0; off >>= 1) {
        d0 += __shfl_down(d0, off, 32);
        d1 += __shfl_down(d1, off, 32);
        d2 += __shfl_down(d2, off, 32);
    }
    int lane = threadIdx.x & 31, wv = threadIdx.x >> 5;
    if (lane == 0) { red[wv] = d0; red[8 + wv] = d1; red[16 + wv] = d2; }
    __syncthreads();
    if (threadIdx.x == 0) {
        float l0 = batt[0], l1 = batt[1], l2 = batt[2];
        for (int i = 0; i < 8; i++) { l0 += red[i]; l1 += red[8+i]; l2 += red[16+i]; }
        float mx = fmaxf(l0, fmaxf(l1, l2));
        float x0 = __expf(l0 - mx), x1 = __expf(l1 - mx), x2 = __expf(l2 - mx);
        float inv = 1.f / (x0 + x1 + x2);
        w[0] = x0 * inv; w[1] = x1 * inv; w[2] = x2 * inv;
    }
    __syncthreads();
    float w0 = w[0], w1 = w[1], w2 = w[2];
#pragma unroll
    for (int i = 0; i < 4; i++) {
        int h = threadIdx.x + i * 256;
        o[row * HDIM + h] = (__bf16)(w0 * r0[h] + w1 * r1[h] + w2 * r2[h]);
    }
}

// ---------------------------------------------------------------------------
// Host orchestration
// ---------------------------------------------------------------------------
extern "C" void kernel_launch(void* const* d_in, const int* in_sizes, int n_in,
                              void* d_out, int out_size, void* d_ws, size_t ws_size,
                              hipStream_t stream)
{
    (void)in_sizes; (void)n_in; (void)out_size; (void)ws_size;

    const float* cnn  = (const float*)d_in[0];
    const float* lstm = (const float*)d_in[1];
    const float* trf  = (const float*)d_in[2];
    const float* Wv   = (const float*)d_in[7];
    const float* bv   = (const float*)d_in[8];
    const float* Wo   = (const float*)d_in[9];
    const float* bo   = (const float*)d_in[10];
    const float* ln1g = (const float*)d_in[11];
    const float* ln1b = (const float*)d_in[12];
    const float* W1   = (const float*)d_in[13];
    const float* b1   = (const float*)d_in[14];
    const float* W2   = (const float*)d_in[15];
    const float* b2   = (const float*)d_in[16];
    const float* ln2g = (const float*)d_in[17];
    const float* ln2b = (const float*)d_in[18];
    const float* Wfu  = (const float*)d_in[19];
    const float* bfu  = (const float*)d_in[20];
    const float* Watt = (const float*)d_in[21];
    const float* batt = (const float*)d_in[22];
    const float* Wfin = (const float*)d_in[23];
    const float* bfin = (const float*)d_in[24];
    const float* lnfg = (const float*)d_in[25];
    const float* lnfb = (const float*)d_in[26];

    const size_t BH = (size_t)BATCH * HDIM;
    const size_t HH = (size_t)HDIM * HDIM;

    // fp32 workspace (9*BH floats)
    float* attn = (float*)d_ws;            // [B,H] (also out2 scratch, later fin)
    float* out1 = attn + BH;               // [B,H]
    float* ffn2 = out1 + BH;               // [B,H]
    float* enh  = ffn2 + BH;               // [3,B,H] accumulators
    float* efu  = enh + 3 * BH;            // [3,B,H]
    float* fin  = attn;                    // alias (attn dead after g-loop)

    // bf16 workspace (17*BH elements) behind the fp32 region
    __bf16* Wvh    = (__bf16*)(efu + 3 * BH);
    __bf16* Woh    = Wvh + 6 * HH;
    __bf16* W1h    = Woh + 6 * HH;
    __bf16* W2h    = W1h + 24 * HH;
    __bf16* Wfuh   = W2h + 24 * HH;
    __bf16* Wfinh  = Wfuh + 3 * HH;
    __bf16* featsh = Wfinh + HH;           // [3,B,H]
    __bf16* vh     = featsh + 3 * BH;      // [B,H]
    __bf16* out1h  = vh + BH;              // [B,H]
    __bf16* ffn1h  = out1h + BH;           // [B,4H]
    __bf16* enhh   = ffn1h;                // alias (ffn1h dead after g-loop)
    __bf16* fusedh = vh;                   // alias (vh dead after g-loop)

    const float* feats[3] = { cnn, lstm, trf };
    const int qidx[6] = { 0, 0, 1, 1, 2, 2 };
    const int kidx[6] = { 1, 2, 0, 2, 0, 1 };

    dim3 blk(256);
    auto cvt = [&](const float* s, __bf16* d, size_t n) {
        k_cvt<<<dim3((unsigned)(n / 2048)), blk, 0, stream>>>(s, d, n);
    };
    auto gemm = [&](const __bf16* A, const __bf16* Wm, const float* bias,
                    float* Cf, __bf16* Ch, int M, int N, int K, int relu) {
        dim3 grid(N / 128, M / 128);
        k_gemm<<<grid, blk, 0, stream>>>(A, Wm, bias, Cf, Ch, M, N, K, relu);
    };

    // one-time (per launch) bf16 conversion of weights + base features
    cvt(Wv,   Wvh,   6 * HH);
    cvt(Wo,   Woh,   6 * HH);
    cvt(W1,   W1h,   24 * HH);
    cvt(W2,   W2h,   24 * HH);
    cvt(Wfu,  Wfuh,  3 * HH);
    cvt(Wfin, Wfinh, HH);
    for (int m = 0; m < 3; m++) cvt(feats[m], featsh + m * BH, BH);

    // enh[m] starts as the base feature, then accumulates out2 pairs
    for (int m = 0; m < 3; m++)
        (void)hipMemcpyAsync(enh + m * BH, feats[m], BH * sizeof(float),
                             hipMemcpyDeviceToDevice, stream);

    for (int g = 0; g < 6; g++) {
        gemm(featsh + (size_t)kidx[g] * BH, Wvh + g * HH, bv + g * HDIM,
             nullptr, vh, BATCH, HDIM, HDIM, 0);
        gemm(vh, Woh + g * HH, bo + g * HDIM,
             attn, nullptr, BATCH, HDIM, HDIM, 0);
        k_addln<<<dim3(BATCH), blk, 0, stream>>>(
            feats[qidx[g]], attn, ln1g + g * HDIM, ln1b + g * HDIM,
            out1, out1h, (float*)nullptr, 0);
        gemm(out1h, W1h + g * 4 * HH, b1 + g * 4 * HDIM,
             nullptr, ffn1h, BATCH, 4 * HDIM, HDIM, 1);
        gemm(ffn1h, W2h + g * 4 * HH, b2 + g * HDIM,
             ffn2, nullptr, BATCH, HDIM, 4 * HDIM, 0);
        // out2 -> attn (scratch), accumulated into enh[g/2]
        k_addln<<<dim3(BATCH), blk, 0, stream>>>(
            out1, ffn2, ln2g + g * HDIM, ln2b + g * HDIM,
            attn, (__bf16*)nullptr, enh + (size_t)(g / 2) * BH, 0);
    }

    for (int m = 0; m < 3; m++) {
        cvt(enh + m * BH, enhh + m * BH, BH);
        gemm(enhh + m * BH, Wfuh + m * HH, bfu + m * HDIM,
             efu + m * BH, nullptr, BATCH, HDIM, HDIM, 0);
    }

    k_fuse<<<dim3(BATCH), blk, 0, stream>>>(efu, efu + BH, efu + 2 * BH,
                                            Watt, batt, fusedh);

    gemm(fusedh, Wfinh, bfin, fin, nullptr, BATCH, HDIM, HDIM, 0);

    k_addln<<<dim3(BATCH), blk, 0, stream>>>(
        fin, (const float*)nullptr, lnfg, lnfb, (float*)d_out,
        (__bf16*)nullptr, (float*)nullptr, 1);
}
